// MnistModel_74113955660226
// MI455X (gfx1250) — matrix operand order repaired and checked by reading, not compiled, not run on hardware
//
#include <hip/hip_runtime.h>
#include <hip/hip_bf16.h>

// ---------------- model dims ----------------
#define HIDDIM 1024
#define HD     64
#define NHEAD  16
#define SEQ    784
#define BATCH  8
#define MTOK   (BATCH * SEQ)     // 6272 = 49 * 128
#define NEXP   8
#define FFDIM  4096
#define NVOCAB 256

typedef __attribute__((ext_vector_type(16))) __bf16 v16bf;
typedef __attribute__((ext_vector_type(8)))  float  v8f;

__device__ __forceinline__ float bf2f(unsigned short h) {
    return __uint_as_float(((unsigned int)h) << 16);
}
__device__ __forceinline__ unsigned short f2bf(float f) {
    unsigned int u = __float_as_uint(f);
    unsigned int r = u + 0x7FFFu + ((u >> 16) & 1u);   // RNE
    return (unsigned short)(r >> 16);
}

// ---- CDNA5 async memory->LDS copy (ASYNCcnt-tracked, no VGPR staging) ----
// vdst = per-lane LDS byte address; for an LDS object, the low 32 bits of the
// generic (flat) address are exactly the LDS offset (aperture rule, ISA 10.2).
__device__ __forceinline__ void async_ld_b128(const void* gptr, void* lptr) {
    unsigned lds_off = (unsigned)(unsigned long long)(uintptr_t)lptr;
    asm volatile("global_load_async_to_lds_b128 %0, %1, off"
                 :: "v"(lds_off), "v"(gptr)
                 : "memory");
}
__device__ __forceinline__ void wait_async0() {
    asm volatile("s_wait_asynccnt 0" ::: "memory");
}

// ---------------- WMMA GEMM: C[M,N] = act(A[M,K] @ Bt[N,K]^T + bias) (* gate) ----------------
#define BM 128
#define BN 128
#define BK 32

template<int GELU, int GATED, int BF16OUT, int ACCUM>
__global__ __launch_bounds__(256) void gemm_bf16_wmma(
    const unsigned short* __restrict__ A,    // M x K, row-major bf16
    const unsigned short* __restrict__ Bt,   // N x K, row-major bf16 (pre-transposed weights)
    const float* __restrict__ bias,          // length N
    const float* __restrict__ gate, int gstride,
    float* __restrict__ Cf, unsigned short* __restrict__ Cb,
    int M, int N, int K)
{
    // double-buffered tiles: 2 * (8KB + 8KB) = 32KB
    __shared__ __align__(16) unsigned short sA[2][BM * BK];
    __shared__ __align__(16) unsigned short sB[2][BN * BK];

    const int tid  = threadIdx.x;
    const int lane = tid & 31;        // wave32
    const int wave = tid >> 5;        // 0..7
    const int wm   = wave & 3;        // 4 waves along M (32 rows each)
    const int wn   = wave >> 2;       // 2 waves along N (64 cols each)
    const int m0   = blockIdx.y * BM;
    const int n0   = blockIdx.x * BN;
    const int lh   = lane >> 4;       // lane half (0/1)
    const int ll   = lane & 15;

    // each thread DMAs 2 x 16B chunks into sA and 2 into sB per tile
    const int r0 = tid >> 2,          c0 = (tid & 3) << 3;
    const int r1 = (tid + 256) >> 2,  c1 = ((tid + 256) & 3) << 3;

    auto issue_tile = [&](int buf, int k0) {
        async_ld_b128(A  + (size_t)(m0 + r0) * K + k0 + c0, &sA[buf][r0 * BK + c0]);
        async_ld_b128(Bt + (size_t)(n0 + r0) * K + k0 + c0, &sB[buf][r0 * BK + c0]);
        async_ld_b128(A  + (size_t)(m0 + r1) * K + k0 + c1, &sA[buf][r1 * BK + c1]);
        async_ld_b128(Bt + (size_t)(n0 + r1) * K + k0 + c1, &sB[buf][r1 * BK + c1]);
    };

    v8f zero8 = {0.f, 0.f, 0.f, 0.f, 0.f, 0.f, 0.f, 0.f};
    v8f acc[2][4];
#pragma unroll
    for (int i = 0; i < 2; ++i)
#pragma unroll
        for (int j = 0; j < 4; ++j) acc[i][j] = zero8;

    const int nk = K / BK;
    issue_tile(0, 0);                       // prologue: prefetch first tile

    for (int it = 0; it < nk; ++it) {
        const int buf = it & 1;
        wait_async0();                      // my async copies for `buf` landed
        __syncthreads();                    // everyone's copies landed; prev reads of buf^1 done
        if (it + 1 < nk) issue_tile(buf ^ 1, (it + 1) * BK);   // overlap next DMA with math

        // preload ALL fragments, then run an uninterrupted WMMA burst
        // A fragments (ISA 16-bit A 16x32 layout): lane half h covers K = h*8+i and 16+h*8+i
        uint4 aLo[2], aHi[2], bLo[4], bHi[4];
#pragma unroll
        for (int i = 0; i < 2; ++i) {
            const unsigned short* pa = &sA[buf][(wm * 32 + i * 16 + ll) * BK];
            aLo[i] = *(const uint4*)(pa + lh * 8);
            aHi[i] = *(const uint4*)(pa + 16 + lh * 8);
        }
#pragma unroll
        for (int j = 0; j < 4; ++j) {
            // B fragment: column (ll), lane half h covers K = h*8+i and 16+h*8+i
            const unsigned short* pb = &sB[buf][(wn * 64 + j * 16 + ll) * BK];
            bLo[j] = *(const uint4*)(pb + lh * 8);
            bHi[j] = *(const uint4*)(pb + 16 + lh * 8);
        }
#pragma unroll
        for (int j = 0; j < 4; ++j) {
            union { uint4 u[2]; v16bf v; } Bf;
            Bf.u[0] = bLo[j]; Bf.u[1] = bHi[j];
#pragma unroll
            for (int i = 0; i < 2; ++i) {
                union { uint4 u[2]; v16bf v; } Af;
                Af.u[0] = aLo[i]; Af.u[1] = aHi[i];
                acc[i][j] = __builtin_amdgcn_wmma_f32_16x16x32_bf16(
                    false, Af.v, false, Bf.v, (short)0, acc[i][j], false, false);
            }
        }
    }

    // epilogue: C layout (ISA): VGPR r -> M = r + lh*8, lane ll -> N
#pragma unroll
    for (int i = 0; i < 2; ++i) {
#pragma unroll
        for (int j = 0; j < 4; ++j) {
#pragma unroll
            for (int r = 0; r < 8; ++r) {
                int m = m0 + wm * 32 + i * 16 + lh * 8 + r;
                int n = n0 + wn * 64 + j * 16 + ll;
                float val = acc[i][j][r] + bias[n];
                if (GELU)  val = 0.5f * val * (1.0f + erff(val * 0.70710678118654752f));
                if (GATED) val *= gate[(size_t)m * gstride];
                size_t o = (size_t)m * N + n;
                if (BF16OUT)    Cb[o] = f2bf(val);
                else if (ACCUM) Cf[o] += val;
                else            Cf[o] = val;
            }
        }
    }
}

// ---------------- elementwise / small kernels ----------------
__global__ void embed_kernel(const float* __restrict__ x, const float* __restrict__ cond,
                             const float* __restrict__ iw, const float* __restrict__ ib,
                             const float* __restrict__ sw, const float* __restrict__ sb,
                             float* __restrict__ xs)
{
    long long idx = (long long)blockIdx.x * blockDim.x + threadIdx.x;
    if (idx >= (long long)MTOK * HIDDIM) return;
    int tok  = (int)(idx >> 10);
    int hcol = (int)(idx & 1023);
    int b = tok / SEQ, n = tok % SEQ;
    float v;
    if (n == 0) v = cond[b] * sw[hcol] + sb[hcol];
    else        v = x[b * SEQ + (n - 1)] * iw[hcol] + ib[hcol];
    xs[idx] = v;
}

__global__ __launch_bounds__(256) void layernorm_bf16_kernel(
    const float* __restrict__ x, const float* __restrict__ g, const float* __restrict__ be,
    unsigned short* __restrict__ y)
{
    __shared__ float sred[256];
    int tok = blockIdx.x, tid = threadIdx.x;
    const float* xr = x + (size_t)tok * HIDDIM;
    float s = 0.f;
    for (int i = tid; i < HIDDIM; i += 256) s += xr[i];
    sred[tid] = s; __syncthreads();
    for (int off = 128; off; off >>= 1) { if (tid < off) sred[tid] += sred[tid + off]; __syncthreads(); }
    float mu = sred[0] * (1.0f / HIDDIM);
    __syncthreads();
    float v = 0.f;
    for (int i = tid; i < HIDDIM; i += 256) { float d = xr[i] - mu; v += d * d; }
    sred[tid] = v; __syncthreads();
    for (int off = 128; off; off >>= 1) { if (tid < off) sred[tid] += sred[tid + off]; __syncthreads(); }
    float inv = rsqrtf(sred[0] * (1.0f / HIDDIM) + 1e-5f);
    unsigned short* yr = y + (size_t)tok * HIDDIM;
    for (int i = tid; i < HIDDIM; i += 256)
        yr[i] = f2bf((xr[i] - mu) * inv * g[i] + be[i]);
}

__global__ __launch_bounds__(256) void router_kernel(
    const unsigned short* __restrict__ x, const float* __restrict__ rw, const float* __restrict__ rb,
    float* __restrict__ gates)
{
    __shared__ float sred[256];
    __shared__ float slog[NEXP];
    int tok = blockIdx.x, tid = threadIdx.x;
    const unsigned short* xr = x + (size_t)tok * HIDDIM;
    float p[NEXP];
#pragma unroll
    for (int e = 0; e < NEXP; ++e) p[e] = 0.f;
    for (int i = tid; i < HIDDIM; i += 256) {
        float xv = bf2f(xr[i]);
        const float* r = rw + (size_t)i * NEXP;
#pragma unroll
        for (int e = 0; e < NEXP; ++e) p[e] += xv * r[e];
    }
    for (int e = 0; e < NEXP; ++e) {
        sred[tid] = p[e]; __syncthreads();
        for (int off = 128; off; off >>= 1) { if (tid < off) sred[tid] += sred[tid + off]; __syncthreads(); }
        if (tid == 0) slog[e] = sred[0] + rb[e];
        __syncthreads();
    }
    if (tid == 0) {
        float mx = slog[0];
        for (int e = 1; e < NEXP; ++e) mx = fmaxf(mx, slog[e]);
        float ex[NEXP];
        for (int e = 0; e < NEXP; ++e) ex[e] = __expf(slog[e] - mx);
        int i1 = 0;
        for (int e = 1; e < NEXP; ++e) if (ex[e] > ex[i1]) i1 = e;
        int i2 = (i1 == 0) ? 1 : 0;
        for (int e = 0; e < NEXP; ++e) if (e != i1 && ex[e] > ex[i2]) i2 = e;
        float tot = ex[i1] + ex[i2];
        float* gr = gates + (size_t)tok * NEXP;
        for (int e = 0; e < NEXP; ++e) gr[e] = 0.f;
        gr[i1] = ex[i1] / tot;
        gr[i2] = ex[i2] / tot;
    }
}

__global__ void rope_kernel(float* __restrict__ q, float* __restrict__ k)
{
    long long idx = (long long)blockIdx.x * blockDim.x + threadIdx.x;  // MTOK*NHEAD*32
    if (idx >= (long long)MTOK * NHEAD * 32) return;
    int d2  = (int)(idx & 31);
    int h   = (int)((idx >> 5) & 15);
    int tok = (int)(idx >> 9);
    int n   = tok % SEQ;
    float dv  = __expf(-(float)d2 * (logf(10000.0f) / 32.0f));
    float ang = (float)n * dv;
    float s = __sinf(ang), c = __cosf(ang);
    size_t o = (size_t)tok * HIDDIM + h * HD + d2;
    float q1 = q[o], q2 = q[o + 32];
    q[o]      = q1 + q1 * c - q2 * s;   // t + (t*cos + rot*sin)
    q[o + 32] = q2 + q2 * c + q1 * s;
    float k1 = k[o], k2 = k[o + 32];
    k[o]      = k1 + k1 * c - k2 * s;
    k[o + 32] = k2 + k2 * c + k1 * s;
}

__global__ __launch_bounds__(256) void attn_kernel(
    const float* __restrict__ Q, const float* __restrict__ K,
    const float* __restrict__ V, float* __restrict__ O)
{
    __shared__ float sS[SEQ];
    __shared__ float sQ[HD];
    __shared__ float sred[256];
    const int qi = blockIdx.x, h = blockIdx.y, b = blockIdx.z;
    const int tid = threadIdx.x;
    if (tid < HD) sQ[tid] = Q[((size_t)(b * SEQ + qi)) * HIDDIM + h * HD + tid];
    __syncthreads();

    float lmax = -1e30f;
    for (int j = tid; j <= qi; j += 256) {
        const float* kr = K + ((size_t)(b * SEQ + j)) * HIDDIM + h * HD;
        float s = 0.f;
#pragma unroll
        for (int d = 0; d < HD; ++d) s += sQ[d] * kr[d];
        s *= 0.125f;                 // HD^-0.5
        sS[j] = s;
        lmax = fmaxf(lmax, s);
    }
    sred[tid] = lmax; __syncthreads();
    for (int off = 128; off; off >>= 1) { if (tid < off) sred[tid] = fmaxf(sred[tid], sred[tid + off]); __syncthreads(); }
    float m = sred[0];
    __syncthreads();
    float lsum = 0.f;
    for (int j = tid; j <= qi; j += 256) { float p = __expf(sS[j] - m); sS[j] = p; lsum += p; }
    sred[tid] = lsum; __syncthreads();
    for (int off = 128; off; off >>= 1) { if (tid < off) sred[tid] += sred[tid + off]; __syncthreads(); }
    float invsum = 1.0f / sred[0];
    __syncthreads();

    const int d = tid & 63, g = tid >> 6;   // 4 key-groups x 64 dims
    float acc = 0.f;
    for (int j = g; j <= qi; j += 4)
        acc += sS[j] * V[((size_t)(b * SEQ + j)) * HIDDIM + h * HD + d];
    sred[tid] = acc; __syncthreads();
    if (g == 0) {
        float r = sred[d] + sred[64 + d] + sred[128 + d] + sred[192 + d];
        O[((size_t)(b * SEQ + qi)) * HIDDIM + h * HD + d] = r * invsum;
    }
}

__global__ void zero_f32_kernel(float* __restrict__ p, long long n)
{
    long long i = (long long)blockIdx.x * blockDim.x + threadIdx.x;
    if (i < n) p[i] = 0.f;
}
__global__ void add_inplace_kernel(float* __restrict__ a, const float* __restrict__ b, long long n)
{
    long long i = (long long)blockIdx.x * blockDim.x + threadIdx.x;
    if (i < n) a[i] += b[i];
}
__global__ void cvt_bf16_kernel(const float* __restrict__ x, unsigned short* __restrict__ y, long long n)
{
    long long i = (long long)blockIdx.x * blockDim.x + threadIdx.x;
    if (i < n) y[i] = f2bf(x[i]);
}
// transpose + convert: w is K x N (f32, row-major) -> wt is N x K (bf16, row-major)
__global__ void cvt_w_t_kernel(const float* __restrict__ w, unsigned short* __restrict__ wt,
                               int K, int N)
{
    long long idx = (long long)blockIdx.x * blockDim.x + threadIdx.x;
    if (idx >= (long long)N * K) return;
    int n = (int)(idx / K);
    int k = (int)(idx % K);
    wt[idx] = f2bf(w[(size_t)k * N + n]);
}

// ---------------- orchestration ----------------
extern "C" void kernel_launch(void* const* d_in, const int* in_sizes, int n_in,
                              void* d_out, int out_size, void* d_ws, size_t ws_size,
                              hipStream_t stream)
{
    (void)in_sizes; (void)n_in; (void)out_size; (void)ws_size;

    const float* x    = (const float*)d_in[0];
    const float* cond = (const float*)d_in[1];
    const float* iw   = (const float*)d_in[2];
    const float* ib   = (const float*)d_in[3];
    const float* sw   = (const float*)d_in[4];
    const float* sb   = (const float*)d_in[5];
    const float* ow   = (const float*)d_in[6];
    const float* ob   = (const float*)d_in[7];
    // block params: 22 tensors each, starting at index 8.
    // 0:qw 1:qb 2:kw 3:kb 4:v.rw 5:v.rb 6:v.w 7:v.b 8:p.rw 9:p.rb 10:p.w 11:p.b
    // 12:f.rw 13:f.rb 14:f.w1 15:f.b1 16:f.w2 17:f.b2 18:g1 19:be1 20:g2 21:be2
    auto BP = [&](int pb, int off) -> const float* { return (const float*)d_in[8 + pb * 22 + off]; };

    char* ws = (char*)d_ws;
    size_t off = 0;
    auto alloc = [&](size_t bytes) -> char* {
        char* p = ws + off;
        off = (off + bytes + 255) & ~(size_t)255;
        return p;
    };

    float* xs   = (float*)alloc((size_t)MTOK * HIDDIM * 4);
    float* qf   = (float*)alloc((size_t)MTOK * HIDDIM * 4);
    float* kf   = (float*)alloc((size_t)MTOK * HIDDIM * 4);
    float* vf   = (float*)alloc((size_t)MTOK * HIDDIM * 4);
    float* t0   = (float*)alloc((size_t)MTOK * HIDDIM * 4);
    float* accb = (float*)alloc((size_t)MTOK * HIDDIM * 4);
    unsigned short* lnb  = (unsigned short*)alloc((size_t)MTOK * HIDDIM * 2);
    unsigned short* hbuf = (unsigned short*)alloc((size_t)MTOK * FFDIM * 2);
    float* gates = (float*)alloc((size_t)MTOK * NEXP * 4);
    unsigned short* owt = (unsigned short*)alloc((size_t)NVOCAB * HIDDIM * 2);
    unsigned short *qwt[2], *kwt[2], *vwt[2], *pwt[2], *f1t[2], *f2t[2];
    for (int pb = 0; pb < 2; ++pb) {
        qwt[pb] = (unsigned short*)alloc((size_t)HIDDIM * HIDDIM * 2);
        kwt[pb] = (unsigned short*)alloc((size_t)HIDDIM * HIDDIM * 2);
        vwt[pb] = (unsigned short*)alloc((size_t)NEXP * HIDDIM * HIDDIM * 2);
        pwt[pb] = (unsigned short*)alloc((size_t)NEXP * HIDDIM * HIDDIM * 2);
        f1t[pb] = (unsigned short*)alloc((size_t)NEXP * HIDDIM * FFDIM * 2);
        f2t[pb] = (unsigned short*)alloc((size_t)NEXP * FFDIM * HIDDIM * 2);
    }

    dim3 blk(256);
    auto cgrid = [](long long n) { return dim3((unsigned)((n + 255) / 256)); };
    const long long NELEM = (long long)MTOK * HIDDIM;

    // ---- weight conversions (deterministic, every call) ----
    cvt_w_t_kernel<<<cgrid((long long)NVOCAB * HIDDIM), blk, 0, stream>>>(ow, owt, HIDDIM, NVOCAB);
    for (int pb = 0; pb < 2; ++pb) {
        cvt_w_t_kernel<<<cgrid((long long)HIDDIM * HIDDIM), blk, 0, stream>>>(BP(pb, 0), qwt[pb], HIDDIM, HIDDIM);
        cvt_w_t_kernel<<<cgrid((long long)HIDDIM * HIDDIM), blk, 0, stream>>>(BP(pb, 2), kwt[pb], HIDDIM, HIDDIM);
        for (int e = 0; e < NEXP; ++e) {
            cvt_w_t_kernel<<<cgrid((long long)HIDDIM * HIDDIM), blk, 0, stream>>>(
                BP(pb, 6) + (size_t)e * HIDDIM * HIDDIM, vwt[pb] + (size_t)e * HIDDIM * HIDDIM, HIDDIM, HIDDIM);
            cvt_w_t_kernel<<<cgrid((long long)HIDDIM * HIDDIM), blk, 0, stream>>>(
                BP(pb, 10) + (size_t)e * HIDDIM * HIDDIM, pwt[pb] + (size_t)e * HIDDIM * HIDDIM, HIDDIM, HIDDIM);
            cvt_w_t_kernel<<<cgrid((long long)HIDDIM * FFDIM), blk, 0, stream>>>(
                BP(pb, 14) + (size_t)e * HIDDIM * FFDIM, f1t[pb] + (size_t)e * HIDDIM * FFDIM, HIDDIM, FFDIM);
            cvt_w_t_kernel<<<cgrid((long long)FFDIM * HIDDIM), blk, 0, stream>>>(
                BP(pb, 16) + (size_t)e * FFDIM * HIDDIM, f2t[pb] + (size_t)e * FFDIM * HIDDIM, FFDIM, HIDDIM);
        }
    }

    // ---- embed + shifted start token ----
    embed_kernel<<<cgrid(NELEM), blk, 0, stream>>>(x, cond, iw, ib, sw, sb, xs);

    dim3 gH(HIDDIM / BN, MTOK / BM);   // N=1024
    dim3 gF(FFDIM / BN, MTOK / BM);    // N=4096
    dim3 gO(NVOCAB / BN, MTOK / BM);   // N=256
    dim3 ga(SEQ, NHEAD, BATCH);

    for (int ib_ = 0; ib_ < 8; ++ib_) {
        int pb = ib_ & 1;

        // ---- attention branch ----
        layernorm_bf16_kernel<<<MTOK, blk, 0, stream>>>(xs, BP(pb, 18), BP(pb, 19), lnb);
        gemm_bf16_wmma<0, 0, 0, 0><<<gH, blk, 0, stream>>>(lnb, qwt[pb], BP(pb, 1), nullptr, 0,
                                                           qf, nullptr, MTOK, HIDDIM, HIDDIM);
        gemm_bf16_wmma<0, 0, 0, 0><<<gH, blk, 0, stream>>>(lnb, kwt[pb], BP(pb, 3), nullptr, 0,
                                                           kf, nullptr, MTOK, HIDDIM, HIDDIM);
        // v = MoE(linear experts), dense-all-experts with gate-weighted accumulate
        router_kernel<<<MTOK, blk, 0, stream>>>(lnb, BP(pb, 4), BP(pb, 5), gates);
        zero_f32_kernel<<<cgrid(NELEM), blk, 0, stream>>>(vf, NELEM);
        for (int e = 0; e < NEXP; ++e)
            gemm_bf16_wmma<0, 1, 0, 1><<<gH, blk, 0, stream>>>(
                lnb, vwt[pb] + (size_t)e * HIDDIM * HIDDIM, BP(pb, 7) + (size_t)e * HIDDIM,
                gates + e, NEXP, vf, nullptr, MTOK, HIDDIM, HIDDIM);
        rope_kernel<<<cgrid((long long)MTOK * NHEAD * 32), blk, 0, stream>>>(qf, kf);
        attn_kernel<<<ga, blk, 0, stream>>>(qf, kf, vf, t0);
        // post-MoE projection of attention output
        cvt_bf16_kernel<<<cgrid(NELEM), blk, 0, stream>>>(t0, lnb, NELEM);
        router_kernel<<<MTOK, blk, 0, stream>>>(lnb, BP(pb, 8), BP(pb, 9), gates);
        zero_f32_kernel<<<cgrid(NELEM), blk, 0, stream>>>(accb, NELEM);
        for (int e = 0; e < NEXP; ++e)
            gemm_bf16_wmma<0, 1, 0, 1><<<gH, blk, 0, stream>>>(
                lnb, pwt[pb] + (size_t)e * HIDDIM * HIDDIM, BP(pb, 11) + (size_t)e * HIDDIM,
                gates + e, NEXP, accb, nullptr, MTOK, HIDDIM, HIDDIM);
        add_inplace_kernel<<<cgrid(NELEM), blk, 0, stream>>>(xs, accb, NELEM);

        // ---- FF MoE branch ----
        layernorm_bf16_kernel<<<MTOK, blk, 0, stream>>>(xs, BP(pb, 20), BP(pb, 21), lnb);
        router_kernel<<<MTOK, blk, 0, stream>>>(lnb, BP(pb, 12), BP(pb, 13), gates);
        zero_f32_kernel<<<cgrid(NELEM), blk, 0, stream>>>(accb, NELEM);
        for (int e = 0; e < NEXP; ++e) {
            gemm_bf16_wmma<1, 0, 1, 0><<<gF, blk, 0, stream>>>(
                lnb, f1t[pb] + (size_t)e * HIDDIM * FFDIM, BP(pb, 15) + (size_t)e * FFDIM,
                nullptr, 0, nullptr, hbuf, MTOK, FFDIM, HIDDIM);
            gemm_bf16_wmma<0, 1, 0, 1><<<gH, blk, 0, stream>>>(
                hbuf, f2t[pb] + (size_t)e * FFDIM * HIDDIM, BP(pb, 17) + (size_t)e * HIDDIM,
                gates + e, NEXP, accb, nullptr, MTOK, HIDDIM, FFDIM);
        }
        add_inplace_kernel<<<cgrid(NELEM), blk, 0, stream>>>(xs, accb, NELEM);
    }

    // ---- output projection ----
    cvt_bf16_kernel<<<cgrid(NELEM), blk, 0, stream>>>(xs, lnb, NELEM);
    gemm_bf16_wmma<0, 0, 0, 0><<<gO, blk, 0, stream>>>(lnb, owt, ob, nullptr, 0,
                                                       (float*)d_out, nullptr, MTOK, NVOCAB, HIDDIM);
}
